// FixedAddressMemoryWrite_2714419331664
// MI455X (gfx1250) — matrix operand order repaired and checked by reading, not compiled, not run on hardware
//
#include <hip/hip_runtime.h>
#include <hip/hip_bf16.h>
#include <math.h>

typedef _Float16 half_t;
typedef __attribute__((ext_vector_type(16))) _Float16 v16h;
typedef __attribute__((ext_vector_type(8)))  _Float16 v8h;
typedef __attribute__((ext_vector_type(8)))  float    v8f;
typedef __attribute__((ext_vector_type(4)))  float    v4f;

#define NB   64
#define NT   4096
#define NS   64
#define NE   64
#define NA   32
#define CHUNKS 16
#define ROWS_PER_BLOCK (NT / CHUNKS)        // 256 t-rows per block
#define PASS_ROWS 64                        // 4 waves * 16 rows
#define NPASS (ROWS_PER_BLOCK / PASS_ROWS)  // 4
#define NWAVE 4
#define SCR_STRIDE 68                       // 64 + pad; keeps 16B alignment

union V16U { v16h v; v8h h[2]; };

__device__ __forceinline__ v8f wmma_f16(v16h a, v16h b, v8f c) {
  // D = A(16x32 f16) x B(32x16 f16) + C(16x16 f32)
  return __builtin_amdgcn_wmma_f32_16x16x32_f16(false, a, false, b, (short)0, c,
                                                false, false);
}

__device__ __forceinline__ void wave_sync() {
  // per-wave LDS ops are in-order in HW; this only pins the compiler's ordering
  asm volatile("" ::: "memory");
  __builtin_amdgcn_wave_barrier();
  asm volatile("" ::: "memory");
}

// xor-shuffle within each 16-lane group via v_permlane16_b32 (VALU, no LDS)
template <unsigned LO, unsigned HI>
__device__ __forceinline__ float permxor(float v) {
  int s = __float_as_int(v);
  int r = __builtin_amdgcn_permlane16(s, s, (int)LO, (int)HI, false, false);
  return __int_as_float(r);
}
#define XOR1(v)  permxor<0x67452301u, 0xEFCDAB89u>(v)
#define XOR2(v)  permxor<0x54761032u, 0xDCFE98BAu>(v)
#define XOR4(v)  permxor<0x32107654u, 0xBA98FEDCu>(v)
#define XOR8(v)  permxor<0xFEDCBA98u, 0x76543210u>(v)

// lane ^ 16 via v_permlanex16_b32 (identity selector into the other row)
__device__ __forceinline__ float permxor16(float v) {
  int s = __float_as_int(v);
  int r = __builtin_amdgcn_permlanex16(s, s, (int)0x76543210, (int)0xFEDCBA98,
                                       false, false);
  return __int_as_float(r);
}

__device__ __forceinline__ v8h cvt8(const float* p) {
  v4f a = *(const v4f*)p;
  v4f b = *(const v4f*)(p + 4);
  v8h r;
  r[0]=(_Float16)a[0]; r[1]=(_Float16)a[1]; r[2]=(_Float16)a[2]; r[3]=(_Float16)a[3];
  r[4]=(_Float16)b[0]; r[5]=(_Float16)b[1]; r[6]=(_Float16)b[2]; r[7]=(_Float16)b[3];
  return r;
}

__device__ __forceinline__ v16h ldB16(const half_t* p) {  // halves p[0..7], p[16..23]
  V16U u;
  u.h[0] = *(const v8h*)p;
  u.h[1] = *(const v8h*)(p + 16);
  return u.v;
}

// ---------------------------------------------------------------- zero ws
__global__ void fam_zero(float* __restrict__ p, int n) {
  int i = blockIdx.x * blockDim.x + threadIdx.x;
  if (i < n) p[i] = 0.0f;
}

// ------------------------------------------------- prep: f16 weights + addr
__global__ void fam_prep(const float* __restrict__ ma, const float* __restrict__ Wq,
                         const float* __restrict__ Wv, const float* __restrict__ Wg,
                         half_t* __restrict__ addrN, half_t* __restrict__ Wq16,
                         half_t* __restrict__ Wv16,  half_t* __restrict__ Wg16) {
  int tid = threadIdx.x;  // one block, 256 threads
  if (tid < NS) {
    float ss = 0.0f;
    for (int a = 0; a < NA; ++a) { float v = ma[tid * NA + a]; ss += v * v; }
    float scl = 1.0f / fmaxf(sqrtf(ss), 1e-12f);
    for (int a = 0; a < NA; ++a)
      addrN[tid * NA + a] = (half_t)(ma[tid * NA + a] * scl);
  }
  for (int i = tid; i < NA * NE; i += 256) Wq16[i] = (half_t)Wq[i];
  for (int i = tid; i < NE * NE; i += 256) Wv16[i] = (half_t)Wv[i];
  if (tid < NE) Wg16[tid] = (half_t)Wg[tid];
}

// ---------------------------------------------------------------- main
__global__ __launch_bounds__(128)
void fam_main(const float* __restrict__ x, const float* __restrict__ bgp,
              float* __restrict__ acc, float* __restrict__ wsum,
              const half_t* __restrict__ addrN, const half_t* __restrict__ Wq16,
              const half_t* __restrict__ Wv16,  const half_t* __restrict__ Wg16) {
  __shared__ __align__(16) half_t wT[NS * PASS_ROWS];        // w transposed [s][t]
  __shared__ __align__(16) half_t vT[NE * PASS_ROWS];        // v transposed [e][t]
  __shared__ __align__(16) half_t qn[NWAVE][16 * NA];        // normalized q rows
  __shared__ __align__(16) float scr[NWAVE][16 * SCR_STRIDE];// per-wave f32 scratch
  __shared__ float gates[NWAVE][16];

  const int tid  = threadIdx.x;
  const int wave = tid >> 5;
  const int lo   = tid & 15;        // lane % 16
  const int hi   = (tid >> 4) & 1;  // lane / 16
  const int kb   = hi * 8;          // B-fragment K base (lane group)
  const int ab   = hi * 8;          // A-fragment K base (lane group)
  const int b    = blockIdx.y;
  const int tstart = blockIdx.x * ROWS_PER_BLOCK;
  const float bg = bgp[0];

  v8f accC[4] = {{}, {}, {}, {}};   // wave owns s-tile=wave, e-tiles 0..3

  for (int p = 0; p < NPASS; ++p) {
    // ---- A-fragment of x (16 rows x K=64 f16), straight from global ----
    const int trow = tstart + p * PASS_ROWS + wave * 16 + lo;
    const float* xr = x + ((size_t)b * NT + trow) * NE;
    V16U a0, a1;
    a0.h[0] = cvt8(xr + ab);       a0.h[1] = cvt8(xr + ab + 16);
    a1.h[0] = cvt8(xr + 32 + ab);  a1.h[1] = cvt8(xr + 32 + ab + 16);
    if (p + 1 < NPASS) __builtin_prefetch(xr + (size_t)PASS_ROWS * NE, 0, 1);

    // ---- gate logits via WMMA: B column 0 = Wg, rest 0 ----
    v16h bgl0 = {}, bgl1 = {};
    if (lo == 0) { bgl0 = ldB16(Wg16 + kb); bgl1 = ldB16(Wg16 + 32 + kb); }
    v8f cg = {};
    cg = wmma_f16(a0.v, bgl0, cg);
    cg = wmma_f16(a1.v, bgl1, cg);
    if (lo == 0) {
      #pragma unroll
      for (int r = 0; r < 8; ++r) {
        float z = cg[r] + bg;
        gates[wave][r + 8 * hi] = 1.0f / (1.0f + __expf(-z));
      }
    }

    // ---- q = x @ Wq^T  [16 x 32] ----
    v8f q0 = {}, q1 = {};
    q0 = wmma_f16(a0.v, ldB16(Wq16 + lo * NE + kb), q0);
    q0 = wmma_f16(a1.v, ldB16(Wq16 + lo * NE + 32 + kb), q0);
    q1 = wmma_f16(a0.v, ldB16(Wq16 + (16 + lo) * NE + kb), q1);
    q1 = wmma_f16(a1.v, ldB16(Wq16 + (16 + lo) * NE + 32 + kb), q1);

    // ---- per-row L2 normalize: VALU permlane reductions over N group ----
    #pragma unroll
    for (int r = 0; r < 8; ++r) {
      float ss = q0[r] * q0[r] + q1[r] * q1[r];
      ss += XOR1(ss);
      ss += XOR2(ss);
      ss += XOR4(ss);
      ss += XOR8(ss);
      float scl = 1.0f / fmaxf(sqrtf(ss), 1e-12f);
      const int m = r + 8 * hi;
      qn[wave][m * NA + lo]      = (half_t)(q0[r] * scl);
      qn[wave][m * NA + 16 + lo] = (half_t)(q1[r] * scl);
    }
    wave_sync();

    // ---- scores = qn @ addrN^T  [16 x 64] ----
    V16U aq;
    aq.h[0] = *(const v8h*)&qn[wave][lo * NA + ab];
    aq.h[1] = *(const v8h*)&qn[wave][lo * NA + ab + 16];
    #pragma unroll
    for (int st = 0; st < 4; ++st) {
      v8f sc = {};
      sc = wmma_f16(aq.v, ldB16(addrN + (st * 16 + lo) * NA + kb), sc);
      #pragma unroll
      for (int r = 0; r < 8; ++r)
        scr[wave][(r + 8 * hi) * SCR_STRIDE + st * 16 + lo] = sc[r];
    }
    wave_sync();

    // ---- softmax over S (x4 temperature), x gate, write wT[s][t] f16 ----
    {
      const v4f* srow4 = (const v4f*)&scr[wave][lo * SCR_STRIDE + hi * 32];
      float l[32];
      float mx = -3.4e38f;
      #pragma unroll
      for (int j = 0; j < 8; ++j) {
        v4f t = srow4[j];
        #pragma unroll
        for (int k = 0; k < 4; ++k) {
          l[4 * j + k] = t[k] * 4.0f;
          mx = fmaxf(mx, l[4 * j + k]);
        }
      }
      mx = fmaxf(mx, permxor16(mx));
      float sum = 0.0f;
      #pragma unroll
      for (int i = 0; i < 32; ++i) { l[i] = __expf(l[i] - mx); sum += l[i]; }
      sum += permxor16(sum);
      const float scl = gates[wave][lo] / sum;
      const int tcol = wave * 16 + lo;
      #pragma unroll
      for (int i = 0; i < 32; ++i)
        wT[(hi * 32 + i) * PASS_ROWS + tcol] = (half_t)(l[i] * scl);
    }
    wave_sync();

    // ---- v = x @ Wv^T  [16 x 64] (through scratch) ----
    #pragma unroll
    for (int et = 0; et < 4; ++et) {
      v8f vc = {};
      vc = wmma_f16(a0.v, ldB16(Wv16 + (et * 16 + lo) * NE + kb), vc);
      vc = wmma_f16(a1.v, ldB16(Wv16 + (et * 16 + lo) * NE + 32 + kb), vc);
      #pragma unroll
      for (int r = 0; r < 8; ++r)
        scr[wave][(r + 8 * hi) * SCR_STRIDE + et * 16 + lo] = vc[r];
    }
    wave_sync();

    // ---- remap scratch -> vT[e][t] f16 ----
    {
      const v4f* srow4 = (const v4f*)&scr[wave][lo * SCR_STRIDE];
      const int tcol = wave * 16 + lo;
      #pragma unroll
      for (int j = 0; j < 8; ++j) {
        v4f t = srow4[hi * 8 + j];
        #pragma unroll
        for (int k = 0; k < 4; ++k) {
          const int e = hi * 32 + 4 * j + k;
          vT[e * PASS_ROWS + tcol] = (half_t)t[k];
        }
      }
    }
    __syncthreads();

    // ---- C[s][e] += w^T @ v, K = PASS_ROWS via K=32 WMMA steps ----
    #pragma unroll
    for (int kc = 0; kc < PASS_ROWS / 32; ++kc) {
      const half_t* wp = &wT[(wave * 16 + lo) * PASS_ROWS + kc * 32];
      V16U aw;
      aw.h[0] = *(const v8h*)(wp + ab);
      aw.h[1] = *(const v8h*)(wp + ab + 16);
      #pragma unroll
      for (int et = 0; et < 4; ++et) {
        const half_t* vp = &vT[(et * 16 + lo) * PASS_ROWS + kc * 32 + kb];
        V16U bv;
        bv.h[0] = *(const v8h*)vp;
        bv.h[1] = *(const v8h*)(vp + 16);
        accC[et] = wmma_f16(aw.v, bv.v, accC[et]);
      }
    }

    // ---- wsum partials: per-s row sums of wT (over t) (vectorized b128 LDS reads) ----
    {
      const int s  = tid & 63;
      const int qh = tid >> 6;
      const v8h* wp8 = (const v8h*)&wT[s * PASS_ROWS + qh * 32];
      float ssum = 0.0f;
      #pragma unroll
      for (int j = 0; j < 4; ++j) {
        v8h w8 = wp8[j];
        #pragma unroll
        for (int k = 0; k < 8; ++k) ssum += (float)w8[k];
      }
      atomicAdd(&wsum[b * NS + s], ssum);
    }
    __syncthreads();
  }

  // ---- flush per-wave f32 accumulators ----
  float* accb = acc + (size_t)b * NS * NE;
  #pragma unroll
  for (int et = 0; et < 4; ++et) {
    #pragma unroll
    for (int r = 0; r < 8; ++r) {
      const int s = wave * 16 + r + 8 * hi;
      atomicAdd(&accb[s * NE + et * 16 + lo], accC[et][r]);
    }
  }
}

// ---------------------------------------------------------------- finalize
__global__ void fam_finalize(const float* __restrict__ mv, const float* __restrict__ acc,
                             const float* __restrict__ wsum, float* __restrict__ out) {
  int idx = blockIdx.x * blockDim.x + threadIdx.x;
  if (idx >= NB * NS * NE) return;
  int bs = idx >> 6;  // b*NS + s
  float sw = wsum[bs];
  float g  = 1.0f - __expf(-sw);
  float upd = acc[idx] / fmaxf(sw, 1e-6f);
  out[idx] = mv[idx] * (1.0f - g) + upd * g;
}

extern "C" void kernel_launch(void* const* d_in, const int* in_sizes, int n_in,
                              void* d_out, int out_size, void* d_ws, size_t ws_size,
                              hipStream_t stream) {
  const float* x  = (const float*)d_in[0];
  const float* ma = (const float*)d_in[1];
  const float* mv = (const float*)d_in[2];
  const float* Wq = (const float*)d_in[3];
  const float* Wv = (const float*)d_in[4];
  const float* Wg = (const float*)d_in[5];
  const float* bg = (const float*)d_in[6];
  float* out = (float*)d_out;

  float*  ws    = (float*)d_ws;
  float*  acc   = ws;                       // NB*NS*NE f32
  float*  wsum  = ws + NB * NS * NE;        // NB*NS f32
  half_t* hb    = (half_t*)(wsum + NB * NS);
  half_t* addrN = hb;                       // NS*NA
  half_t* Wq16  = addrN + NS * NA;          // NA*NE
  half_t* Wv16  = Wq16 + NA * NE;           // NE*NE
  half_t* Wg16  = Wv16 + NE * NE;           // NE

  const int nzero = NB * NS * NE + NB * NS;
  fam_zero<<<(nzero + 255) / 256, 256, 0, stream>>>(acc, nzero);
  fam_prep<<<1, 256, 0, stream>>>(ma, Wq, Wv, Wg, addrN, Wq16, Wv16, Wg16);
  fam_main<<<dim3(CHUNKS, NB), 128, 0, stream>>>(x, bg, acc, wsum,
                                                 addrN, Wq16, Wv16, Wg16);
  fam_finalize<<<(NB * NS * NE + 255) / 256, 256, 0, stream>>>(mv, acc, wsum, out);
  (void)in_sizes; (void)n_in; (void)out_size; (void)ws_size;
}
